// EGNN_51067161149952
// MI455X (gfx1250) — matrix operand order repaired and checked by reading, not compiled, not run on hardware
//
#include <hip/hip_runtime.h>

typedef __attribute__((ext_vector_type(16))) _Float16 v16h;
typedef __attribute__((ext_vector_type(8)))  _Float16 v8h;
typedef __attribute__((ext_vector_type(8)))  float    v8f;

#define HDIM 128
#define XDIM 16
#define PDIM 3

// ---------------------------------------------------------------------------
// WMMA fragment helpers (CDNA5 ISA 7.12.2 layouts, wave32)
// A (16x32 f16): lane<16 holds row M=lane, K {0..7,16..23}; lane>=16 K {8..15,24..31}
// B (32x16 f16, col-major source): lane<16 holds col N=lane, K {0..7,16..23}; lane>=16 K {8..15,24..31}
// ---------------------------------------------------------------------------
static __device__ inline v16h load_a_frag(const _Float16* row, int hi) {
  const _Float16* p = row + (hi ? 8 : 0);
  v8h lo = *(const v8h*)(p);
  v8h hv = *(const v8h*)(p + 16);
  v16h r;
#pragma unroll
  for (int i = 0; i < 8; ++i) { r[i] = lo[i]; r[i + 8] = hv[i]; }
  return r;
}

static __device__ inline v16h load_b_frag(const _Float16* col, int hi) {
  const _Float16* p = col + (hi ? 8 : 0);
  v8h lo = *(const v8h*)(p);
  v8h hv = *(const v8h*)(p + 16);
  v16h r;
#pragma unroll
  for (int i = 0; i < 8; ++i) { r[i] = lo[i]; r[i + 8] = hv[i]; }
  return r;
}

static __device__ inline v8f wmma16(v16h a, v16h b, v8f c) {
  // D = A x B + C, f32 accumulate
  return __builtin_amdgcn_wmma_f32_16x16x32_f16(false, a, false, b, (short)0, c,
                                                false, false);
}

// ---------------------------------------------------------------------------
// Small utility kernels
// ---------------------------------------------------------------------------
__global__ __launch_bounds__(256) void egnn_zero_f32(float* __restrict__ p, int n) {
  int i = blockIdx.x * 256 + threadIdx.x;
  if (i < n) p[i] = 0.f;
}

__global__ __launch_bounds__(256) void egnn_f32_to_f16(const float* __restrict__ s,
                                                       _Float16* __restrict__ d, int n) {
  int i = blockIdx.x * 256 + threadIdx.x;
  if (i < n) d[i] = (_Float16)s[i];
}

// dst[n*K + k] = src[k*Nn + n]  (row-major [K][Nn] fp32 -> col-major f16 [Nn][K])
__global__ __launch_bounds__(256) void egnn_w_transpose_f16(
    const float* __restrict__ src, _Float16* __restrict__ dst, int K, int Nn) {
  int i = blockIdx.x * 256 + threadIdx.x;
  if (i >= K * Nn) return;
  int k = i % K, n = i / K;
  dst[(size_t)n * K + k] = (_Float16)src[(size_t)k * Nn + n];
}

// h = concat(x, pos) @ W_enc + b_enc  (K=19, plain FMA is fine: 0.24 GFLOP)
__global__ __launch_bounds__(256) void egnn_encoder(
    const float* __restrict__ x, const float* __restrict__ pos,
    const float* __restrict__ W, const float* __restrict__ b,
    float* __restrict__ h, int N) {
  int idx = blockIdx.x * 256 + threadIdx.x;
  if (idx >= N * HDIM) return;
  int node = idx >> 7, ch = idx & (HDIM - 1);
  float s = b[ch];
  const float* xr = x + (size_t)node * XDIM;
#pragma unroll
  for (int k = 0; k < XDIM; ++k) s += xr[k] * W[k * HDIM + ch];
  const float* pr = pos + (size_t)node * PDIM;
#pragma unroll
  for (int k = 0; k < PDIM; ++k) s += pr[k] * W[(XDIM + k) * HDIM + ch];
  h[idx] = s;
}

__global__ __launch_bounds__(256) void egnn_radial(
    const float* __restrict__ pos, const int* __restrict__ erow,
    const int* __restrict__ ecol, float* __restrict__ rad, int E) {
  int e = blockIdx.x * 256 + threadIdx.x;
  if (e >= E) return;
  const float* a = pos + (size_t)erow[e] * PDIM;
  const float* c = pos + (size_t)ecol[e] * PDIM;
  float s = 0.f;
#pragma unroll
  for (int k = 0; k < PDIM; ++k) { float d = a[k] - c[k]; s += d * d; }
  rad[e] = s;
}

// ---------------------------------------------------------------------------
// Edge MLP (fused): ef = relu(relu([h[row],h[col],radial]@We1+be1)@We2+be2)
// then agg[row] += ef  (global_atomic_add_f32). One wave handles 16 edges.
// ---------------------------------------------------------------------------
__global__ __launch_bounds__(256) void egnn_edge_mlp(
    const _Float16* __restrict__ h16, const float* __restrict__ radial,
    const int* __restrict__ erow, const int* __restrict__ ecol,
    const _Float16* __restrict__ w1,      // [128][256] col-major f16 (We1 rows 0..255)
    const float* __restrict__ w1last,     // We1 row 256 (radial weights), fp32 [128]
    const float* __restrict__ b1, const _Float16* __restrict__ w2,  // [128][128] col-major
    const float* __restrict__ b2, float* __restrict__ agg, int E) {
  __shared__ __align__(32) _Float16 stage[8][16][HDIM];  // per-wave A-staging, 32 KB
  const int lane = threadIdx.x & 31;
  const int wid  = threadIdx.x >> 5;
  const int base = (blockIdx.x * 8 + wid) * 16;
  if (base >= E) return;
  const int m16 = lane & 15;
  const int hi  = lane >> 4;

  int eA = base + m16; if (eA >= E) eA = E - 1;
  const _Float16* rowR = h16 + (size_t)erow[eA] * HDIM;
  const _Float16* rowC = h16 + (size_t)ecol[eA] * HDIM;

  float rad[8];
#pragma unroll
  for (int j = 0; j < 8; ++j) {
    int e = base + j + hi * 8;
    rad[j] = (e < E) ? radial[e] : 0.f;
  }

  // ---- GEMM1: [16x256] @ [256x128] -> 8 C-tiles of 16x16 f32
  v8f acc[8] = {};
#pragma unroll
  for (int kc = 0; kc < 8; ++kc) {
    const _Float16* src = (kc < 4) ? (rowR + kc * 32) : (rowC + (kc - 4) * 32);
    v16h a = load_a_frag(src, hi);
#pragma unroll
    for (int t = 0; t < 8; ++t) {
      v16h b = load_b_frag(w1 + (size_t)(t * 16 + m16) * 256 + kc * 32, hi);
      acc[t] = wmma16(a, b, acc[t]);
    }
  }

  // ---- epilogue1: + bias + radial rank-1, ReLU, restage as f16 A-tile
#pragma unroll
  for (int t = 0; t < 8; ++t) {
    const int n = t * 16 + m16;
    const float bb = b1[n];
    const float wl = w1last[n];
#pragma unroll
    for (int j = 0; j < 8; ++j) {
      float v = acc[t][j] + bb + rad[j] * wl;
      v = v > 0.f ? v : 0.f;
      stage[wid][j + hi * 8][n] = (_Float16)v;  // C layout: (m=j+8*hi, n)
    }
  }

  // ---- GEMM2: [16x128] @ [128x128]
  v8f acc2[8] = {};
#pragma unroll
  for (int kc = 0; kc < 4; ++kc) {
    v16h a = load_a_frag(&stage[wid][m16][kc * 32], hi);
#pragma unroll
    for (int t = 0; t < 8; ++t) {
      v16h b = load_b_frag(w2 + (size_t)(t * 16 + m16) * 128 + kc * 32, hi);
      acc2[t] = wmma16(a, b, acc2[t]);
    }
  }

  // ---- epilogue2: bias + ReLU + segment_sum scatter (f32 atomics)
  int nodes[8];
#pragma unroll
  for (int j = 0; j < 8; ++j) {
    int e = base + j + hi * 8;
    nodes[j] = (e < E) ? erow[e] : -1;
  }
#pragma unroll
  for (int t = 0; t < 8; ++t) {
    const int n = t * 16 + m16;
    const float bb = b2[n];
#pragma unroll
    for (int j = 0; j < 8; ++j) {
      if (nodes[j] >= 0) {
        float v = acc2[t][j] + bb;
        v = v > 0.f ? v : 0.f;
        atomicAdd(agg + (size_t)nodes[j] * HDIM + n, v);
      }
    }
  }
}

// ---------------------------------------------------------------------------
// Node MLP: out = relu([h,agg]@Wn1+bn1)@Wn2+bn2 ; one wave handles 16 nodes
// ---------------------------------------------------------------------------
__global__ __launch_bounds__(256) void egnn_node_mlp(
    const _Float16* __restrict__ h16, const _Float16* __restrict__ agg16,
    const _Float16* __restrict__ w1,   // [128][256] col-major f16
    const float* __restrict__ b1, const _Float16* __restrict__ w2,  // [128][128]
    const float* __restrict__ b2, float* __restrict__ outb, int N) {
  __shared__ __align__(32) _Float16 stage[8][16][HDIM];
  const int lane = threadIdx.x & 31;
  const int wid  = threadIdx.x >> 5;
  const int base = (blockIdx.x * 8 + wid) * 16;
  if (base >= N) return;
  const int m16 = lane & 15;
  const int hi  = lane >> 4;

  int nA = base + m16; if (nA >= N) nA = N - 1;
  const _Float16* rowH = h16   + (size_t)nA * HDIM;
  const _Float16* rowG = agg16 + (size_t)nA * HDIM;

  v8f acc[8] = {};
#pragma unroll
  for (int kc = 0; kc < 8; ++kc) {
    const _Float16* src = (kc < 4) ? (rowH + kc * 32) : (rowG + (kc - 4) * 32);
    v16h a = load_a_frag(src, hi);
#pragma unroll
    for (int t = 0; t < 8; ++t) {
      v16h b = load_b_frag(w1 + (size_t)(t * 16 + m16) * 256 + kc * 32, hi);
      acc[t] = wmma16(a, b, acc[t]);
    }
  }

#pragma unroll
  for (int t = 0; t < 8; ++t) {
    const int n = t * 16 + m16;
    const float bb = b1[n];
#pragma unroll
    for (int j = 0; j < 8; ++j) {
      float v = acc[t][j] + bb;
      v = v > 0.f ? v : 0.f;
      stage[wid][j + hi * 8][n] = (_Float16)v;
    }
  }

  v8f acc2[8] = {};
#pragma unroll
  for (int kc = 0; kc < 4; ++kc) {
    v16h a = load_a_frag(&stage[wid][m16][kc * 32], hi);
#pragma unroll
    for (int t = 0; t < 8; ++t) {
      v16h b = load_b_frag(w2 + (size_t)(t * 16 + m16) * 128 + kc * 32, hi);
      acc2[t] = wmma16(a, b, acc2[t]);
    }
  }

#pragma unroll
  for (int t = 0; t < 8; ++t) {
    const int n = t * 16 + m16;
    const float bb = b2[n];
#pragma unroll
    for (int j = 0; j < 8; ++j) {
      int node = base + j + hi * 8;
      if (node < N) outb[(size_t)node * HDIM + n] = acc2[t][j] + bb;
    }
  }
}

// ---------------------------------------------------------------------------
// LayerNorm + ReLU + residual: h = relu(LN(out)*gamma+beta) + h (in place on h)
// One wave per node; __shfl_xor reduction over wave32.
// ---------------------------------------------------------------------------
__global__ __launch_bounds__(256) void egnn_layernorm(
    const float* __restrict__ outb, const float* __restrict__ gam,
    const float* __restrict__ bet, float* __restrict__ h, int N) {
  const int lane = threadIdx.x & 31;
  const int wid  = threadIdx.x >> 5;
  const int node = blockIdx.x * 8 + wid;
  if (node >= N) return;
  const float* o = outb + (size_t)node * HDIM;
  float v[4], s = 0.f, s2 = 0.f;
#pragma unroll
  for (int i = 0; i < 4; ++i) {
    v[i] = o[lane + 32 * i];
    s += v[i];
    s2 += v[i] * v[i];
  }
#pragma unroll
  for (int off = 16; off >= 1; off >>= 1) {
    s  += __shfl_xor(s, off, 32);
    s2 += __shfl_xor(s2, off, 32);
  }
  const float mu  = s * (1.f / HDIM);
  const float var = s2 * (1.f / HDIM) - mu * mu;
  const float rs  = rsqrtf(var + 1e-5f);
  float* hp = h + (size_t)node * HDIM;
#pragma unroll
  for (int i = 0; i < 4; ++i) {
    int c = lane + 32 * i;
    float y = (v[i] - mu) * rs * gam[c] + bet[c];
    y = y > 0.f ? y : 0.f;
    hp[c] = y + hp[c];  // residual
  }
}

// ---------------------------------------------------------------------------
extern "C" void kernel_launch(void* const* d_in, const int* in_sizes, int n_in,
                              void* d_out, int out_size, void* d_ws, size_t ws_size,
                              hipStream_t stream) {
  (void)n_in; (void)out_size; (void)ws_size;
  const float* x    = (const float*)d_in[0];
  const float* pos  = (const float*)d_in[1];
  const int*   ei   = (const int*)d_in[2];
  // d_in[3] = batch (unused)
  const float* Wenc = (const float*)d_in[4];
  const float* benc = (const float*)d_in[5];
  const float* We1  = (const float*)d_in[6];
  const float* be1  = (const float*)d_in[7];
  const float* We2  = (const float*)d_in[8];
  const float* be2  = (const float*)d_in[9];
  const float* Wn1  = (const float*)d_in[10];
  const float* bn1  = (const float*)d_in[11];
  const float* Wn2  = (const float*)d_in[12];
  const float* bn2  = (const float*)d_in[13];
  const float* gam  = (const float*)d_in[14];
  const float* bet  = (const float*)d_in[15];

  const int N = in_sizes[0] / XDIM;
  const int E = in_sizes[2] / 2;
  const int L = in_sizes[7] / HDIM;
  const int NH = N * HDIM;
  const int* erow = ei;
  const int* ecol = ei + E;
  float* h = (float*)d_out;   // h lives in d_out; final h is the result

  // workspace carve (256B aligned)
  char* ws = (char*)d_ws;
  size_t off = 0;
  auto carve = [&](size_t bytes) -> char* {
    char* p = ws + off;
    off = (off + bytes + 255) & ~(size_t)255;
    return p;
  };
  _Float16* h16   = (_Float16*)carve((size_t)NH * 2);
  float*    agg   = (float*)   carve((size_t)NH * 4);
  _Float16* agg16 = (_Float16*)carve((size_t)NH * 2);
  float*    outb  = (float*)   carve((size_t)NH * 4);
  float*    rad   = (float*)   carve((size_t)E * 4);
  _Float16* we1h  = (_Float16*)carve((size_t)L * 128 * 256 * 2);
  _Float16* we2h  = (_Float16*)carve((size_t)L * 128 * 128 * 2);
  _Float16* wn1h  = (_Float16*)carve((size_t)L * 128 * 256 * 2);
  _Float16* wn2h  = (_Float16*)carve((size_t)L * 128 * 128 * 2);

  const int g256   = (256 * 128 + 255) / 256;  // K=256 transpose grid
  const int g128   = (128 * 128 + 255) / 256;  // K=128 transpose grid
  const int gNH    = (NH + 255) / 256;
  const int gE     = (E + 255) / 256;
  const int gEdge  = ((E + 15) / 16 + 7) / 8;  // 8 tiles (waves) per block
  const int gNode  = ((N + 15) / 16 + 7) / 8;
  const int gLN    = (N + 7) / 8;

  // one-time weight conversion to f16 col-major
  for (int l = 0; l < L; ++l) {
    egnn_w_transpose_f16<<<g256, 256, 0, stream>>>(We1 + (size_t)l * 257 * 128,
                                                   we1h + (size_t)l * 128 * 256, 256, 128);
    egnn_w_transpose_f16<<<g128, 256, 0, stream>>>(We2 + (size_t)l * 128 * 128,
                                                   we2h + (size_t)l * 128 * 128, 128, 128);
    egnn_w_transpose_f16<<<g256, 256, 0, stream>>>(Wn1 + (size_t)l * 256 * 128,
                                                   wn1h + (size_t)l * 128 * 256, 256, 128);
    egnn_w_transpose_f16<<<g128, 256, 0, stream>>>(Wn2 + (size_t)l * 128 * 128,
                                                   wn2h + (size_t)l * 128 * 128, 128, 128);
  }

  egnn_encoder<<<gNH, 256, 0, stream>>>(x, pos, Wenc, benc, h, N);
  egnn_radial<<<gE, 256, 0, stream>>>(pos, erow, ecol, rad, E);

  for (int l = 0; l < L; ++l) {
    egnn_f32_to_f16<<<gNH, 256, 0, stream>>>(h, h16, NH);
    egnn_zero_f32<<<gNH, 256, 0, stream>>>(agg, NH);
    egnn_edge_mlp<<<gEdge, 256, 0, stream>>>(
        h16, rad, erow, ecol,
        we1h + (size_t)l * 128 * 256,
        We1 + (size_t)l * 257 * 128 + 256 * 128,   // radial row of We1, fp32
        be1 + (size_t)l * HDIM,
        we2h + (size_t)l * 128 * 128,
        be2 + (size_t)l * HDIM, agg, E);
    egnn_f32_to_f16<<<gNH, 256, 0, stream>>>(agg, agg16, NH);
    egnn_node_mlp<<<gNode, 256, 0, stream>>>(
        h16, agg16,
        wn1h + (size_t)l * 128 * 256, bn1 + (size_t)l * HDIM,
        wn2h + (size_t)l * 128 * 128, bn2 + (size_t)l * HDIM, outb, N);
    egnn_layernorm<<<gLN, 256, 0, stream>>>(outb, gam + (size_t)l * HDIM,
                                            bet + (size_t)l * HDIM, h, N);
  }
}
